// TemporalAttentionLayer_48395691491597
// MI455X (gfx1250) — compile-verified, matrix operand order checked by reading, not run on hardware
//
#include <hip/hip_runtime.h>

typedef __attribute__((ext_vector_type(16))) _Float16 v16h;
typedef __attribute__((ext_vector_type(8)))  _Float16 v8h;
typedef __attribute__((ext_vector_type(4)))  _Float16 v4h;
typedef __attribute__((ext_vector_type(8)))  float    v8f;
typedef __attribute__((ext_vector_type(4)))  float    v4f;
typedef __attribute__((ext_vector_type(4)))  int      v4i;

#if defined(__HIP_DEVICE_COMPILE__) && __has_builtin(__builtin_amdgcn_global_load_async_to_lds_b128)
#define USE_ASYNC_LDS 1
typedef __attribute__((address_space(1))) v4i* gptr_b128;  // global int4*
typedef __attribute__((address_space(3))) v4i* lptr_b128;  // LDS int4*
#endif

namespace {
constexpr int kB = 4, kN = 1024, kT = 64, kD = 128, kH = 8, kDH = 16;
constexpr int kThreads = 256;   // 8 waves of 32 on gfx1250
constexpr int kWaves = 8;

// LDS strides in halves (strides keep 16B alignment; +8 pad rotates banks)
constexpr int LDT   = kD + 8;   // 136 -> 272B rows
constexpr int LDQK  = kH * 32;  // 256 -> head-strided, zero-padded K
constexpr int LDVT  = kT + 8;   // 72  -> 144B rows
constexpr int LDATT = kT;       // 64  -> 128B rows

constexpr size_t OFF_TF   = 0;                               // float [64][128]
constexpr size_t SZ_TF    = size_t(kT) * kD * 4;             // 32768
constexpr size_t OFF_TH   = OFF_TF + SZ_TF;                  // f16 [64][136]
constexpr size_t SZ_TH    = size_t(kT) * LDT * 2;            // 17408
constexpr size_t OFF_W    = OFF_TH + SZ_TH;                  // f16 4x[128][136]
constexpr size_t SZ_W1    = size_t(kD) * LDT * 2;            // 34816 each
constexpr size_t SZ_W     = 4 * SZ_W1;                       // 139264
constexpr size_t OFF_Q    = OFF_W + SZ_W;                    // f16 [64][256]
constexpr size_t SZ_QK    = size_t(kT) * LDQK * 2;           // 32768
constexpr size_t OFF_K    = OFF_Q + SZ_QK;
constexpr size_t OFF_VT   = OFF_K + SZ_QK;                   // f16 [8][16][72]
constexpr size_t SZ_VT    = size_t(kH) * kDH * LDVT * 2;     // 18432
constexpr size_t OFF_O    = OFF_VT + SZ_VT;                  // f16 [64][136]
constexpr size_t SZ_O     = SZ_TH;
constexpr size_t OFF_ATT  = OFF_O + SZ_O;                    // f16 [8 waves][16][64]
constexpr size_t SZ_ATT   = size_t(kWaves) * 16 * LDATT * 2; // 16384
constexpr size_t OFF_BIAS = OFF_ATT + SZ_ATT;                // float [128]
constexpr size_t SZ_BIAS  = kD * 4;
constexpr size_t SMEM_BYTES = OFF_BIAS + SZ_BIAS;            // 307712 B < 320 KB
} // namespace

// ---- weight prep: pack Wq^T, Wk^T, Wv^T, lin_w into d_ws as f16 [128][128] ----
__global__ __launch_bounds__(kThreads)
void prep_weights_kernel(const float* __restrict__ gWq, const float* __restrict__ gWk,
                         const float* __restrict__ gWv, const float* __restrict__ gLw,
                         _Float16* __restrict__ ws) {
  const int idx = blockIdx.x * kThreads + threadIdx.x;   // 0 .. 16383
  if (idx < kD * kD) {
    const int d = idx >> 7, e = idx & 127;
    ws[0 * kD * kD + e * kD + d] = (_Float16)gWq[idx];   // B stored N-major
    ws[1 * kD * kD + e * kD + d] = (_Float16)gWk[idx];
    ws[2 * kD * kD + e * kD + d] = (_Float16)gWv[idx];
    ws[3 * kD * kD + idx]        = (_Float16)gLw[idx];   // lin_w already [out][in]
  }
}

// Load one 16x32 f16 WMMA A/B fragment from row-major (N-major for B) LDS.
// ISA layout: lanes 0-15 row r0+lane hold K k0+0..7 and k0+16..23;
//             lanes 16-31 hold K k0+8..15 and k0+24..31.
__device__ __forceinline__ v16h ldsLoadFrag(const _Float16* __restrict__ base,
                                            int ld, int r0, int k0) {
  const int lane = threadIdx.x & 31;
  const int hi   = lane >> 4;
  const int r    = lane & 15;
  const _Float16* p = base + (size_t)(r0 + r) * ld + k0 + hi * 8;
  v8h a = *(const v8h*)p;
  v8h b = *(const v8h*)(p + 16);
  v16h out;
#pragma unroll
  for (int i = 0; i < 8; ++i) { out[i] = a[i]; out[i + 8] = b[i]; }
  return out;
}

__device__ __forceinline__ void waitAsyncZero() {
#ifdef USE_ASYNC_LDS
#if __has_builtin(__builtin_amdgcn_s_wait_asynccnt)
  __builtin_amdgcn_s_wait_asynccnt(0);
#else
  asm volatile("s_wait_asynccnt 0" ::: "memory");
#endif
#endif
}

// Copy one pre-packed f16 [128][128] weight matrix into a padded LDS panel,
// using the CDNA5 async global->LDS path when available (tracked by ASYNCcnt).
__device__ __forceinline__ void copyWeightPanel(const _Float16* __restrict__ src,
                                                _Float16* __restrict__ dst, int tid) {
#pragma unroll
  for (int ii = 0; ii < (kD * kD / 8) / kThreads; ++ii) {   // 2048 b128 chunks
    const int i = tid + ii * kThreads;
    const int row = i >> 4, c = (i & 15) << 3;
#ifdef USE_ASYNC_LDS
    __builtin_amdgcn_global_load_async_to_lds_b128(
        (gptr_b128)(src + (size_t)row * kD + c),
        (lptr_b128)(dst + (size_t)row * LDT + c),
        0, 0);
#else
    *(v8h*)(dst + (size_t)row * LDT + c) = *(const v8h*)(src + (size_t)row * kD + c);
#endif
  }
}

// One projection: [64,128] @ panel -> Q/K (head-padded) or V (transposed per head)
template <int MAT>
__device__ __forceinline__ void projectQKV(const _Float16* __restrict__ sTh,
                                           const _Float16* __restrict__ sW,
                                           _Float16* __restrict__ sQ,
                                           _Float16* __restrict__ sK,
                                           _Float16* __restrict__ sVt,
                                           int wave, int lane) {
  const int hiHalf = lane >> 4, lcol = lane & 15;
#pragma unroll
  for (int jj = 0; jj < 4; ++jj) {
    const int j = wave + jj * kWaves;      // mt = jj, nt = wave
    const int mt = j >> 3, nt = j & 7;
    v8f acc = {};
#pragma unroll
    for (int ks = 0; ks < 4; ++ks) {
      v16h a = ldsLoadFrag(sTh, LDT, mt * 16, ks * 32);
      v16h b = ldsLoadFrag(sW,  LDT, nt * 16, ks * 32);
      acc = __builtin_amdgcn_wmma_f32_16x16x32_f16(false, a, false, b,
                                                   (short)0, acc, false, false);
    }
    const int h = nt;                      // 16-wide N tile == one head
    const int row0 = mt * 16 + hiHalf * 8;
    if constexpr (MAT == 2) {
      v8h o;                               // sVt[h][d][t]: 8 contiguous t per lane
#pragma unroll
      for (int r = 0; r < 8; ++r) o[r] = (_Float16)acc[r];
      *(v8h*)(sVt + ((size_t)h * kDH + lcol) * LDVT + row0) = o;
    } else {
      _Float16* dst = (MAT == 0) ? sQ : sK;
#pragma unroll
      for (int r = 0; r < 8; ++r)
        dst[(size_t)(row0 + r) * LDQK + h * 32 + lcol] = (_Float16)acc[r];
    }
  }
}

__global__ __launch_bounds__(kThreads)
void temporal_attn_kernel(const float* __restrict__ gIn,
                          const float* __restrict__ gPos,
                          const _Float16* __restrict__ gWs,  // prepacked f16 weights
                          const float* __restrict__ gLb,
                          float* __restrict__ gOut) {
  extern __shared__ char smem[];
  float*    sTf  = (float*)(smem + OFF_TF);
  _Float16* sTh  = (_Float16*)(smem + OFF_TH);
  _Float16* sWq  = (_Float16*)(smem + OFF_W);
  _Float16* sWk  = sWq + (size_t)kD * LDT;
  _Float16* sWv  = sWk + (size_t)kD * LDT;
  _Float16* sWl  = sWv + (size_t)kD * LDT;
  _Float16* sQ   = (_Float16*)(smem + OFF_Q);
  _Float16* sK   = (_Float16*)(smem + OFF_K);
  _Float16* sVt  = (_Float16*)(smem + OFF_VT);
  _Float16* sO   = (_Float16*)(smem + OFF_O);
  _Float16* sAtt = (_Float16*)(smem + OFF_ATT);
  float*    sBia = (float*)(smem + OFF_BIAS);

  const int tid    = threadIdx.x;
  const int wave   = tid >> 5;
  const int lane   = tid & 31;
  const int hiHalf = lane >> 4;
  const int lcol   = lane & 15;
  const size_t tileBase = (size_t)blockIdx.x * (kT * kD);

  // Kick off weight panel fills first (async path overlaps with phase 1).
  copyWeightPanel(gWs + 0 * kD * kD, sWq, tid);
  copyWeightPanel(gWs + 1 * kD * kD, sWk, tid);
  copyWeightPanel(gWs + 2 * kD * kD, sWv, tid);
  copyWeightPanel(gWs + 3 * kD * kD, sWl, tid);

  // ---- Phase 1: temporal = inputs + pos_emb (fp32 residual copy + f16 copy) ----
#pragma unroll
  for (int ii = 0; ii < (kT * kD / 4) / kThreads; ++ii) {
    const int i = tid + ii * kThreads;
    v4f a = *(const v4f*)(gIn + tileBase + (size_t)i * 4);
    v4f p = *(const v4f*)(gPos + (size_t)i * 4);   // pos_emb [64][128] matches tile
    v4f v = a + p;
    *(v4f*)(sTf + (size_t)i * 4) = v;
    v4h hv;
#pragma unroll
    for (int q = 0; q < 4; ++q) hv[q] = (_Float16)v[q];
    const int t = (i * 4) >> 7, d = (i * 4) & 127;
    *(v4h*)(sTh + (size_t)t * LDT + d) = hv;
  }
  { // zero head-pad halves of sQ/sK once (K-padding trick: DH=16 -> K=32 WMMA)
    v8h z = {};
#pragma unroll
    for (int ii = 0; ii < (kT * LDQK / 8) / kThreads; ++ii) {
      const int i = tid + ii * kThreads;
      ((v8h*)sQ)[i] = z;
      ((v8h*)sK)[i] = z;
    }
  }
  if (tid < kD) sBia[tid] = gLb[tid];

  waitAsyncZero();       // weight panels resident in LDS
  __syncthreads();

  // ---- Phase 2: Q/K/V projections, straight-line per matrix ----
  projectQKV<0>(sTh, sWq, sQ, sK, sVt, wave, lane);
  projectQKV<1>(sTh, sWk, sQ, sK, sVt, wave, lane);
  projectQKV<2>(sTh, sWv, sQ, sK, sVt, wave, lane);
  __syncthreads();

  // ---- Phase 3: causal attention per (head, 16-row block) ----
  _Float16* sAttW = sAtt + (size_t)wave * 16 * LDATT;   // wave-private scratch
#pragma unroll
  for (int jj = 0; jj < 4; ++jj) {
    const int j = wave + jj * kWaves;
    const int h = j >> 2, mt = j & 3;
    // scores[16 x 64] = qh @ kh^T (K=32, zero-padded upper half)
    v16h aq = ldsLoadFrag(sQ + h * 32, LDQK, mt * 16, 0);
    v8f s[4];
#pragma unroll
    for (int st = 0; st < 4; ++st) {
      v16h bk = ldsLoadFrag(sK + h * 32, LDQK, st * 16, 0);
      v8f z = {};
      s[st] = __builtin_amdgcn_wmma_f32_16x16x32_f16(false, aq, false, bk,
                                                     (short)0, z, false, false);
    }
    // scale, causal mask, in-register softmax (a row lives in one 16-lane half)
    const float scale = 0.25f;   // 1/sqrt(DH)
#pragma unroll
    for (int r = 0; r < 8; ++r) {
      const int trow = mt * 16 + hiHalf * 8 + r;
      float v0[4];
#pragma unroll
      for (int st = 0; st < 4; ++st) {
        const int scol = st * 16 + lcol;
        v0[st] = (scol <= trow) ? s[st][r] * scale : -1.0e30f;
      }
      float m = fmaxf(fmaxf(v0[0], v0[1]), fmaxf(v0[2], v0[3]));
#pragma unroll
      for (int off = 8; off >= 1; off >>= 1) m = fmaxf(m, __shfl_xor(m, off, 32));
      float p[4], sum = 0.f;
#pragma unroll
      for (int st = 0; st < 4; ++st) { p[st] = __expf(v0[st] - m); sum += p[st]; }
#pragma unroll
      for (int off = 8; off >= 1; off >>= 1) sum += __shfl_xor(sum, off, 32);
      const float inv = 1.0f / sum;
      const int arow = hiHalf * 8 + r;
#pragma unroll
      for (int st = 0; st < 4; ++st)
        sAttW[(size_t)arow * LDATT + st * 16 + lcol] = (_Float16)(p[st] * inv);
    }
    asm volatile("" ::: "memory");  // LDS ops are in-order per wave (DScnt)
    // attn[16x64] @ vh[64x16] -> attn_out tile [16x16]
    v8f oacc = {};
#pragma unroll
    for (int kt = 0; kt < 2; ++kt) {
      v16h a2 = ldsLoadFrag(sAttW, LDATT, 0, kt * 32);
      v16h b2 = ldsLoadFrag(sVt + (size_t)h * kDH * LDVT, LDVT, 0, kt * 32);
      oacc = __builtin_amdgcn_wmma_f32_16x16x32_f16(false, a2, false, b2,
                                                    (short)0, oacc, false, false);
    }
    const int row0 = mt * 16 + hiHalf * 8;
#pragma unroll
    for (int r = 0; r < 8; ++r)
      sO[(size_t)(row0 + r) * LDT + h * 16 + lcol] = (_Float16)oacc[r];
  }
  __syncthreads();

  // ---- Phase 4: FF = relu(attn_out @ lin_w^T + b) + attn_out + temporal ----
#pragma unroll
  for (int jj = 0; jj < 4; ++jj) {
    const int j = wave + jj * kWaves;
    const int mt = j >> 3, nt = j & 7;
    v8f acc = {};
#pragma unroll
    for (int ks = 0; ks < 4; ++ks) {
      v16h a = ldsLoadFrag(sO,  LDT, mt * 16, ks * 32);
      v16h b = ldsLoadFrag(sWl, LDT, nt * 16, ks * 32);
      acc = __builtin_amdgcn_wmma_f32_16x16x32_f16(false, a, false, b,
                                                   (short)0, acc, false, false);
    }
    const int col  = nt * 16 + lcol;
    const float bias = sBia[col];
    const int row0 = mt * 16 + hiHalf * 8;
#pragma unroll
    for (int r = 0; r < 8; ++r) {
      const int row = row0 + r;
      const float ff = fmaxf(acc[r] + bias, 0.0f);
      const float res = ff + (float)sO[(size_t)row * LDT + col] + sTf[row * kD + col];
      gOut[tileBase + (size_t)row * kD + col] = res;
    }
  }
}

extern "C" void kernel_launch(void* const* d_in, const int* in_sizes, int n_in,
                              void* d_out, int out_size, void* d_ws, size_t ws_size,
                              hipStream_t stream) {
  const float* gIn  = (const float*)d_in[0];
  const float* gPos = (const float*)d_in[1];
  const float* gWq  = (const float*)d_in[2];
  const float* gWk  = (const float*)d_in[3];
  const float* gWv  = (const float*)d_in[4];
  const float* gLw  = (const float*)d_in[5];
  const float* gLb  = (const float*)d_in[6];
  float* gOut = (float*)d_out;
  _Float16* ws = (_Float16*)d_ws;   // 4 * 128*128 f16 = 128 KB

  // 1) pack weights once (f16, pre-transposed for N-major WMMA B operands)
  prep_weights_kernel<<<dim3((kD * kD + kThreads - 1) / kThreads), dim3(kThreads),
                        0, stream>>>(gWq, gWk, gWv, gLw, ws);
  // 2) fused attention layer, one workgroup per (b, n) site
  temporal_attn_kernel<<<dim3(kB * kN), dim3(kThreads), SMEM_BYTES, stream>>>(
      gIn, gPos, ws, gLb, gOut);
  (void)in_sizes; (void)n_in; (void)out_size; (void)ws_size;
}
